// CE_41532333752407
// MI455X (gfx1250) — matrix operand order repaired and checked by reading, not compiled, not run on hardware
//
#include <hip/hip_runtime.h>

// ---------------- constants ----------------
#define BB   128      // batch
#define TT   200      // time steps
#define EE   300      // embedding dim
#define HH   512      // hidden
#define G4   2048     // 4*H
#define KPAD 832      // 512 (h) + 320 (x padded from 300)
#define LDA  832      // row stride (elements) of A=[h|x] and Wc

typedef __attribute__((ext_vector_type(16))) __bf16 v16bf;
typedef __attribute__((ext_vector_type(8)))  float  v8f;

union FragU { v16bf v; uint4 q[2]; };

// float -> bf16 (round to nearest even), stored as ushort
__device__ __forceinline__ unsigned short f2bf(float f) {
    unsigned int u = __float_as_uint(f);
    u = (u + 0x7FFFu + ((u >> 16) & 1u)) >> 16;
    return (unsigned short)u;
}

// A fragment (16x32 bf16, MxK): lane<16 -> row=lane, K = {k0..k0+7, k0+16..k0+23}
//                               lane>=16 -> row=lane-16, K shifted by +8
__device__ __forceinline__ v16bf loadA(const unsigned short* p, int ld, int lane) {
    const int r  = lane & 15;
    const int ko = (lane >> 4) * 8;
    const unsigned short* row = p + r * ld;
    FragU f;
    f.q[0] = *(const uint4*)(row + ko);        // K = ko .. ko+7
    f.q[1] = *(const uint4*)(row + ko + 16);   // K = ko+16 .. ko+23
    return f.v;
}

// B fragment (32x16 bf16, KxN), weights stored as N rows x K cols (row-major):
// lane<16 -> col N=lane, K = {k0..k0+7, k0+16..k0+23} ; lane>=16 -> col N=lane-16, K shifted by +8
__device__ __forceinline__ v16bf loadB(const unsigned short* __restrict__ p,
                                       int ld, int lane) {
    const int n  = lane & 15;
    const int ko = (lane >> 4) * 8;
    const unsigned short* row = p + n * ld;
    FragU f;
    f.q[0] = *(const uint4*)(row + ko);        // K = ko .. ko+7
    f.q[1] = *(const uint4*)(row + ko + 16);   // K = ko+16 .. ko+23
    return f.v;
}

__device__ __forceinline__ float sigf(float x) { return 1.0f / (1.0f + __expf(-x)); }

// ---------------- prep: Wc = bf16([W_hh | W_ih | 0pad]), bias = bih + bhh ----
__global__ void prep_w(unsigned short* __restrict__ Wc, float* __restrict__ bias,
                       const float* __restrict__ Wih, const float* __restrict__ Whh,
                       const float* __restrict__ bih, const float* __restrict__ bhh) {
    const int n = blockIdx.x;                     // 0..2047 gate-output row
    for (int k = threadIdx.x; k < KPAD; k += blockDim.x) {
        float v;
        if (k < HH)            v = Whh[n * HH + k];
        else if (k < HH + EE)  v = Wih[n * EE + (k - HH)];
        else                   v = 0.0f;
        Wc[n * LDA + k] = f2bf(v);
    }
    if (threadIdx.x == 0) bias[n] = bih[n] + bhh[n];
}

// ---------------- init: zero h-part of A, set c (opt from src), zero mh/mc ---
__global__ void lstm_init(unsigned short* A, float* c, float* mh, float* mc,
                          const float* c_src) {
    const int idx = blockIdx.x * blockDim.x + threadIdx.x;  // 0 .. B*H-1
    const int b = idx >> 9;          // /512
    const int j = idx & 511;
    A[b * LDA + j] = 0;              // bf16 zero for h part
    const float ci = c_src ? c_src[idx] : 0.0f;
    c[idx]  = ci;
    mh[idx] = 0.0f;
    mc[idx] = 0.0f;
}

// ---------------- gather x_t = bf16(emb[tokens[:,t]]) into A[:,512:832] -----
__global__ void gather_x(unsigned short* __restrict__ A,
                         const int* __restrict__ tokens,
                         const float* __restrict__ emb, int t) {
    const int b = blockIdx.x;        // 0..127
    const int e = threadIdx.x;       // 0..319
    const int tok = tokens[b * TT + t];
    const float v = (e < EE) ? emb[tok * EE + e] : 0.0f;
    A[b * LDA + HH + e] = f2bf(v);
}

// ---------------- one LSTM timestep -----------------------------------------
// grid = 128 blocks (8 batch-tiles x 16 hidden col-pair tiles), block = 128.
// Wave w computes gate-w (i/f/g/o) tiles at N = w*512 + j_tile*32 (+0, +16):
// two independent WMMA accumulation chains sharing one A fragment.
// The 16x832 bf16 A slab is staged into LDS once per block with the gfx1250
// async-to-LDS path (ASYNCcnt), then read back as WMMA fragments via DS.
__global__ void lstm_step(const unsigned short* __restrict__ Acur,
                          unsigned short* __restrict__ Anxt,
                          const unsigned short* __restrict__ Wc,
                          const float* __restrict__ bias,
                          float* __restrict__ c,
                          float* __restrict__ mh,
                          float* __restrict__ mc,
                          const int* __restrict__ tokens,
                          int t) {
    __shared__ unsigned short shA[16 * LDA];       // 26,624 B staged [h|x] slab
    __shared__ float gl[4][2][8][32];              // i/f/g/o x 2 subtiles, fp32

    const int tid    = threadIdx.x;
    const int lane   = tid & 31;
    const int wave   = tid >> 5;
    const int m_tile = blockIdx.x >> 4;            // 0..7  (batch tile)
    const int j_tile = blockIdx.x & 15;            // 0..15 (32-wide col tile)

    // ---- async copy of the contiguous 26,624-byte A slab into LDS ----
    {
        const char*  gsrc  = (const char*)(Acur + (m_tile * 16) * LDA);
        unsigned int lbase = (unsigned int)(size_t)(&shA[0]);
        #pragma unroll
        for (int i = 0; i < 13; ++i) {             // 13 * 128 lanes * 16 B
            const int    off    = (i * 128 + tid) * 16;
            unsigned int ldsoff = lbase + off;
            const char*  ga     = gsrc + off;
            asm volatile("global_load_async_to_lds_b128 %0, %1, off"
                         :: "v"(ldsoff), "v"(ga) : "memory");
        }
        asm volatile("s_wait_asynccnt 0x0" ::: "memory");
    }
    __syncthreads();

    const int Nbase = wave * HH + j_tile * 32;     // gate column base
    const unsigned short* B0 = Wc + (size_t)Nbase * LDA;
    const unsigned short* B1 = Wc + (size_t)(Nbase + 16) * LDA;

    v8f acc0 = {}, acc1 = {};
    #pragma unroll 2
    for (int k0 = 0; k0 < KPAD; k0 += 32) {
        v16bf a  = loadA(shA + k0, LDA, lane);     // ds_load_b128 from LDS
        v16bf b0 = loadB(B0 + k0, LDA, lane);
        v16bf b1 = loadB(B1 + k0, LDA, lane);
        acc0 = __builtin_amdgcn_wmma_f32_16x16x32_bf16(
                   false, a, false, b0, (short)0, acc0, false, false);
        acc1 = __builtin_amdgcn_wmma_f32_16x16x32_bf16(
                   false, a, false, b1, (short)0, acc1, false, false);
    }

    const float bn0 = bias[Nbase + (lane & 15)];
    const float bn1 = bias[Nbase + 16 + (lane & 15)];
    #pragma unroll
    for (int r = 0; r < 8; ++r) {
        gl[wave][0][r][lane] = acc0[r] + bn0;
        gl[wave][1][r][lane] = acc1[r] + bn1;
    }
    __syncthreads();

    // ---- cell/hidden update + mask blend: 512 elements over 128 threads ----
    #pragma unroll
    for (int e = 0; e < 4; ++e) {
        const int idx = e * 128 + tid;             // 0..511
        const int l2  = idx & 31;
        const int r   = (idx >> 5) & 7;
        const int s   = idx >> 8;                  // subtile 0/1
        const float ig = gl[0][s][r][l2];
        const float fg = gl[1][s][r][l2];
        const float gg = gl[2][s][r][l2];
        const float og = gl[3][s][r][l2];
        const int m    = r + 8 * (l2 >> 4);        // row within 16x16 C tile
        const int brow = m_tile * 16 + m;          // global batch index
        const int j    = j_tile * 32 + s * 16 + (l2 & 15);
        const float c_old = c[brow * HH + j];
        const float cn = sigf(fg) * c_old + sigf(ig) * tanhf(gg);
        const float hn = sigf(og) * tanhf(cn);
        c[brow * HH + j]     = cn;
        Anxt[brow * LDA + j] = f2bf(hn);           // h feeds next step's GEMM
        if (tokens[brow * TT + t] != 0) {          // mask blend (m in {0,1})
            mh[brow * HH + j] = hn;
            mc[brow * HH + j] = cn;
        }
    }
}

// ---------------- final: feat=[mh | sim] @ fc_W^T + fc_b, log_softmax -------
__global__ void final_fc(const float* __restrict__ mh,
                         const float* __restrict__ sim,
                         const float* __restrict__ fcW,
                         const float* __restrict__ fcb,
                         float* __restrict__ out) {
    const int b = threadIdx.x;       // 0..127
    float lg[3];
    #pragma unroll
    for (int cc = 0; cc < 3; ++cc) {
        const float* wrow = fcW + cc * (HH + 1);
        float s = fcb[cc];
        for (int j = 0; j < HH; ++j) s += wrow[j] * mh[b * HH + j];
        s += wrow[HH] * sim[b];
        lg[cc] = s;
    }
    const float mx  = fmaxf(lg[0], fmaxf(lg[1], lg[2]));
    const float se  = __expf(lg[0] - mx) + __expf(lg[1] - mx) + __expf(lg[2] - mx);
    const float lse = __logf(se) + mx;
    out[b * 3 + 0] = lg[0] - lse;
    out[b * 3 + 1] = lg[1] - lse;
    out[b * 3 + 2] = lg[2] - lse;
}

// ---------------- host orchestration ----------------------------------------
extern "C" void kernel_launch(void* const* d_in, const int* in_sizes, int n_in,
                              void* d_out, int out_size, void* d_ws, size_t ws_size,
                              hipStream_t stream) {
    const int*   premise    = (const int*)  d_in[0];
    const int*   hypothesis = (const int*)  d_in[1];
    const float* similarity = (const float*)d_in[2];
    const float* emb        = (const float*)d_in[3];
    const float* Wih_p      = (const float*)d_in[4];
    const float* Whh_p      = (const float*)d_in[5];
    const float* bih_p      = (const float*)d_in[6];
    const float* bhh_p      = (const float*)d_in[7];
    const float* Wih_h      = (const float*)d_in[8];
    const float* Whh_h      = (const float*)d_in[9];
    const float* bih_h      = (const float*)d_in[10];
    const float* bhh_h      = (const float*)d_in[11];
    const float* fc_W       = (const float*)d_in[12];
    const float* fc_b       = (const float*)d_in[13];
    float* out = (float*)d_out;

    char* w = (char*)d_ws;
    auto carve = [&](size_t bytes) {
        void* p = (void*)w;
        w += (bytes + 255) & ~(size_t)255;
        return p;
    };
    unsigned short* A0    = (unsigned short*)carve((size_t)BB * LDA * 2);
    unsigned short* A1    = (unsigned short*)carve((size_t)BB * LDA * 2);
    float*          cbuf  = (float*)carve((size_t)BB * HH * 4);
    float*          mh    = (float*)carve((size_t)BB * HH * 4);
    float*          mc    = (float*)carve((size_t)BB * HH * 4);
    unsigned short* WcP   = (unsigned short*)carve((size_t)G4 * LDA * 2);
    unsigned short* WcH   = (unsigned short*)carve((size_t)G4 * LDA * 2);
    float*          biasP = (float*)carve((size_t)G4 * 4);
    float*          biasH = (float*)carve((size_t)G4 * 4);

    // one-time weight packing (bf16, combined [W_hh | W_ih])
    prep_w<<<G4, 256, 0, stream>>>(WcP, biasP, Wih_p, Whh_p, bih_p, bhh_p);
    prep_w<<<G4, 256, 0, stream>>>(WcH, biasH, Wih_h, Whh_h, bih_h, bhh_h);

    // ---- premise LSTM: h0=c0=0 ----
    lstm_init<<<256, 256, 0, stream>>>(A0, cbuf, mh, mc, nullptr);
    gather_x<<<BB, 320, 0, stream>>>(A0, premise, emb, 0);
    for (int t = 0; t < TT; ++t) {
        unsigned short* cur = (t & 1) ? A1 : A0;
        unsigned short* nxt = (t & 1) ? A0 : A1;
        if (t + 1 < TT)
            gather_x<<<BB, 320, 0, stream>>>(nxt, premise, emb, t + 1);
        lstm_step<<<128, 128, 0, stream>>>(cur, nxt, WcP, biasP,
                                           cbuf, mh, mc, premise, t);
    }

    // ---- hypothesis LSTM: h0=0, c0 = masked final c of premise (mc) ----
    lstm_init<<<256, 256, 0, stream>>>(A0, cbuf, mh, mc, mc);
    gather_x<<<BB, 320, 0, stream>>>(A0, hypothesis, emb, 0);
    for (int t = 0; t < TT; ++t) {
        unsigned short* cur = (t & 1) ? A1 : A0;
        unsigned short* nxt = (t & 1) ? A0 : A1;
        if (t + 1 < TT)
            gather_x<<<BB, 320, 0, stream>>>(nxt, hypothesis, emb, t + 1);
        lstm_step<<<128, 128, 0, stream>>>(cur, nxt, WcH, biasH,
                                           cbuf, mh, mc, hypothesis, t);
    }

    // ---- classifier head ----
    final_fc<<<1, BB, 0, stream>>>(mh, similarity, fc_W, fc_b, out);
}
